// GFR2MultiHeadLatentAttention_46351287058504
// MI455X (gfx1250) — matrix operand order repaired and checked by reading, not compiled, not run on hardware
//
#include <hip/hip_runtime.h>

// ---------------------------------------------------------------------------
// MLA (DeepSeek-style multi-head latent attention) forward for MI455X/gfx1250.
// All matmuls routed through v_wmma_f32_16x16x32_bf16 (fp32 inputs converted
// to bf16 on the fly, fp32 accumulate).  Wave32 layouts per CDNA5 ISA 7.12.2.
// v2: double-buffered LDS GEMM pipeline, 4x2 wave tiling (32x64 per wave),
//     batched fragment loads before each WMMA chain, global_prefetch_b8 hints.
// ---------------------------------------------------------------------------

constexpr int kB   = 2;
constexpr int kS   = 2048;
constexpr int kD   = 2048;
constexpr int kH   = 16;
constexpr int kDN  = 128;
constexpr int kDR  = 64;
constexpr int kDV  = 128;
constexpr int kQL  = 1536;
constexpr int kKVL = 512;
constexpr int kDQ  = kDN + kDR;     // 192
constexpr int kM   = kB * kS;       // 4096 token rows
constexpr float kEps = 1e-6f;

typedef __attribute__((ext_vector_type(16))) __bf16 v16bf;
typedef __attribute__((ext_vector_type(8)))  __bf16 v8bf;
typedef __attribute__((ext_vector_type(8)))  float  v8f;

union Frag16 { v16bf v; v8bf h[2]; };

__device__ __forceinline__ v8f v8f_zero() {
  v8f x;
#pragma unroll
  for (int i = 0; i < 8; ++i) x[i] = 0.f;
  return x;
}

// ---------------------------------------------------------------------------
// Generic GEMM: C[M,N] = A[M,K] (f32, row-major) x B[K,N] (f32, row-major).
// 128x128 block tile, BK=32, 8 waves arranged 4(M) x 2(N): 32x64 per wave.
// Double-buffered LDS: global loads for tile t+1 overlap WMMAs of tile t.
// M % 128 == 0 and K % 32 == 0 at all call sites; N may be ragged (576).
// ---------------------------------------------------------------------------
constexpr int GBM = 128, GBN = 128, GBK = 32, GLD = 40; // GLD: padded LDS row (bf16)

__global__ __launch_bounds__(256)
void gemm_f32_wmma_bf16(const float* __restrict__ A, const float* __restrict__ Bm,
                        float* __restrict__ C, int N, int K) {
  __shared__ __bf16 Al[2][GBM * GLD];   // [m][k]
  __shared__ __bf16 Bl[2][GBN * GLD];   // [n][k]  (transposed during staging)
  const int tid  = threadIdx.x;
  const int wave = tid >> 5;
  const int lane = tid & 31;
  const int hi   = lane >> 4;
  const int lo   = lane & 15;
  const int bn0  = blockIdx.x * GBN;
  const int bm0  = blockIdx.y * GBM;

  const int ar = tid >> 3;        // A stage: row within pass (0..31)
  const int ac = (tid & 7) * 4;   //          col (0..28 step 4)
  const int bk = tid >> 5;        // B stage: k within pass (0..7)
  const int bc = (tid & 31) * 4;  //          n (0..124 step 4)

  float4 aR[4], bR[4];            // staging registers (global -> LDS)

  auto ldA = [&](int kt) {
#pragma unroll
    for (int p = 0; p < 4; ++p)
      aR[p] = *(const float4*)(A + (size_t)(bm0 + ar + p * 32) * K + kt * GBK + ac);
  };
  auto ldB = [&](int kt) {
    const int nc = bn0 + bc;
#pragma unroll
    for (int p = 0; p < 4; ++p) {
      const size_t base = (size_t)(kt * GBK + bk + p * 8) * N;
      if (nc + 4 <= N) {
        bR[p] = *(const float4*)(Bm + base + nc);
      } else {
        bR[p].x = (nc + 0 < N) ? Bm[base + nc + 0] : 0.f;
        bR[p].y = (nc + 1 < N) ? Bm[base + nc + 1] : 0.f;
        bR[p].z = (nc + 2 < N) ? Bm[base + nc + 2] : 0.f;
        bR[p].w = (nc + 3 < N) ? Bm[base + nc + 3] : 0.f;
      }
    }
  };
  auto stA = [&](int buf) {
#pragma unroll
    for (int p = 0; p < 4; ++p) {
      __bf16* d = &Al[buf][(ar + p * 32) * GLD + ac];
      d[0] = (__bf16)aR[p].x; d[1] = (__bf16)aR[p].y;
      d[2] = (__bf16)aR[p].z; d[3] = (__bf16)aR[p].w;
    }
  };
  auto stB = [&](int buf) {
#pragma unroll
    for (int p = 0; p < 4; ++p) {
      const int kk = bk + p * 8;
      Bl[buf][(bc + 0) * GLD + kk] = (__bf16)bR[p].x;
      Bl[buf][(bc + 1) * GLD + kk] = (__bf16)bR[p].y;
      Bl[buf][(bc + 2) * GLD + kk] = (__bf16)bR[p].z;
      Bl[buf][(bc + 3) * GLD + kk] = (__bf16)bR[p].w;
    }
  };

  v8f acc[2][4];
#pragma unroll
  for (int mt = 0; mt < 2; ++mt)
#pragma unroll
    for (int nt = 0; nt < 4; ++nt) acc[mt][nt] = v8f_zero();

  const int nkt = K / GBK;
  ldA(0); ldB(0); stA(0); stB(0);

  const int mbase = (wave >> 1) * 32;   // 4 wave rows
  const int nbase = (wave & 1) * 64;    // 2 wave cols

  for (int kt = 0; kt < nkt; ++kt) {
    __syncthreads();                    // buf[kt&1] fully staged
    const int cur = kt & 1;
    if (kt + 1 < nkt) { ldA(kt + 1); ldB(kt + 1); }  // overlap with compute
    if (kt + 2 < nkt) {                               // pull t+2 toward L2
      __builtin_prefetch(A + (size_t)(bm0 + ar) * K + (kt + 2) * GBK + ac, 0, 1);
      __builtin_prefetch(Bm + (size_t)((kt + 2) * GBK + bk) * N + bn0 + bc, 0, 1);
    }

    // batched fragment loads, then the WMMA chain
    Frag16 af[2];
#pragma unroll
    for (int mt = 0; mt < 2; ++mt) {
      const int am = mbase + mt * 16 + lo;
      af[mt].h[0] = *(const v8bf*)&Al[cur][am * GLD + 8 * hi];      // k = 8*hi + e
      af[mt].h[1] = *(const v8bf*)&Al[cur][am * GLD + 16 + 8 * hi]; // k = 16+8*hi+e
    }
    Frag16 bfr[4];
#pragma unroll
    for (int nt = 0; nt < 4; ++nt) {
      const int bn = nbase + nt * 16 + lo;
      bfr[nt].h[0] = *(const v8bf*)&Bl[cur][bn * GLD + 8 * hi];     // k = 8*hi + e
      bfr[nt].h[1] = *(const v8bf*)&Bl[cur][bn * GLD + 16 + 8 * hi];
    }
#pragma unroll
    for (int mt = 0; mt < 2; ++mt)
#pragma unroll
      for (int nt = 0; nt < 4; ++nt)
        acc[mt][nt] = __builtin_amdgcn_wmma_f32_16x16x32_bf16(
            false, af[mt].v, false, bfr[nt].v, (short)0, acc[mt][nt], false, false);

    if (kt + 1 < nkt) { stA(cur ^ 1); stB(cur ^ 1); }  // stage next buffer
  }

  // C layout per 16x16 tile: m = r + 8*hi, n = lo
#pragma unroll
  for (int mt = 0; mt < 2; ++mt) {
    const int mb = bm0 + mbase + mt * 16 + 8 * hi;
#pragma unroll
    for (int nt = 0; nt < 4; ++nt) {
      const int nc = bn0 + nbase + nt * 16 + lo;
      if (nc < N) {
#pragma unroll
        for (int r = 0; r < 8; ++r)
          C[(size_t)(mb + r) * N + nc] = acc[mt][nt][r];
      }
    }
  }
}

// ---------------------------------------------------------------------------
// Row RMSNorm (f32 -> f32), one block per row.
// ---------------------------------------------------------------------------
__global__ __launch_bounds__(256)
void rmsnorm_rows(const float* __restrict__ X, const float* __restrict__ W,
                  float* __restrict__ Y, int ncols, int in_stride, int out_stride) {
  const int row = blockIdx.x;
  const float* xr = X + (size_t)row * in_stride;
  float ss = 0.f;
  for (int c = threadIdx.x; c < ncols; c += 256) { const float v = xr[c]; ss = fmaf(v, v, ss); }
#pragma unroll
  for (int off = 16; off > 0; off >>= 1) ss += __shfl_xor(ss, off, 32);
  __shared__ float red[8];
  __shared__ float tot;
  if ((threadIdx.x & 31) == 0) red[threadIdx.x >> 5] = ss;
  __syncthreads();
  if (threadIdx.x < 32) {
    float v = (threadIdx.x < 8) ? red[threadIdx.x] : 0.f;
#pragma unroll
    for (int off = 4; off > 0; off >>= 1) v += __shfl_xor(v, off, 32);
    if (threadIdx.x == 0) tot = v;
  }
  __syncthreads();
  const float inv = rsqrtf(tot / (float)ncols + kEps);
  float* yr = Y + (size_t)row * out_stride;
  for (int c = threadIdx.x; c < ncols; c += 256) yr[c] = W[c] * xr[c] * inv;
}

// ---------------------------------------------------------------------------
// RoPE + pack Q (scaled by DQ^-1/2), K, and V (transposed [b,h,dv,s]) as bf16.
// One block per token row (b*S + s).
// ---------------------------------------------------------------------------
__global__ __launch_bounds__(256)
void prep_qkv(const float* __restrict__ Qp,   // [M, H*DQ]
              const float* __restrict__ KVp,  // [M, H*(DN+DV)]
              const float* __restrict__ CKV,  // [M, KVL+DR] (k_rot in last DR)
              const float* __restrict__ Cs, const float* __restrict__ Sn,
              __bf16* __restrict__ Qb, __bf16* __restrict__ Kb,
              __bf16* __restrict__ Vt) {
  const int row = blockIdx.x;
  const int b = row / kS, s = row % kS;
  __shared__ float cs[kDR], sn[kDR], krot[kDR];
  if (threadIdx.x < kDR) {
    cs[threadIdx.x] = Cs[(size_t)row * kDR + threadIdx.x];
    sn[threadIdx.x] = Sn[(size_t)row * kDR + threadIdx.x];
  }
  __syncthreads();
  if (threadIdx.x < kDR) {
    const int i = threadIdx.x;
    const float* kr = CKV + (size_t)row * (kKVL + kDR) + kKVL;
    const float x  = kr[i];
    const float xh = (i < kDR / 2) ? -kr[i + kDR / 2] : kr[i - kDR / 2];
    krot[i] = x * cs[i] + xh * sn[i];
  }
  __syncthreads();
  const float scale = rsqrtf((float)kDQ);
  for (int idx = threadIdx.x; idx < kH * kDQ; idx += 256) {
    const int h = idx / kDQ, d = idx % kDQ;
    const float* qr = Qp + (size_t)row * (kH * kDQ) + h * kDQ;
    float qv;
    if (d < kDN) {
      qv = qr[d];
    } else {
      const int i = d - kDN;
      const float x  = qr[kDN + i];
      const float xh = (i < kDR / 2) ? -qr[kDN + i + kDR / 2] : qr[kDN + i - kDR / 2];
      qv = x * cs[i] + xh * sn[i];
    }
    const size_t o = (((size_t)(b * kH + h)) * kS + s) * kDQ + d;
    Qb[o] = (__bf16)(qv * scale);
    const float kvv = (d < kDN)
        ? KVp[(size_t)row * (kH * (kDN + kDV)) + h * (kDN + kDV) + d]
        : krot[d - kDN];
    Kb[o] = (__bf16)kvv;
  }
  for (int idx = threadIdx.x; idx < kH * kDV; idx += 256) {
    const int h = idx / kDV, dv = idx % kDV;
    const float vv = KVp[(size_t)row * (kH * (kDN + kDV)) + h * (kDN + kDV) + kDN + dv];
    Vt[(((size_t)(b * kH + h)) * kDV + dv) * kS + s] = (__bf16)vv;
  }
}

// ---------------------------------------------------------------------------
// Causal flash attention.  One wave per 16-row q tile; 32-key k tiles.
// QK^T: 12 wmma; P.V: 8 wmma per k tile.  Online softmax, f32 state.
// Fragment loads batched ahead of each WMMA chain; next tile prefetched.
// Output written directly in [b, s, h*DV] layout (f32) for the final GEMM.
// ---------------------------------------------------------------------------
constexpr int kDQC = kDQ / 32;   // 6 K-chunks of 32 over the 192-dim head

__global__ __launch_bounds__(256)
void mla_flash(const __bf16* __restrict__ Qb, const __bf16* __restrict__ Kb,
               const __bf16* __restrict__ Vt, float* __restrict__ Ao) {
  __shared__ __bf16 Pl[8][16 * GLD];          // per-wave P staging (layout swap)
  const int wave = threadIdx.x >> 5, lane = threadIdx.x & 31;
  const int hi = lane >> 4, lo = lane & 15;
  const int tile = blockIdx.x * 8 + wave;      // B*H*(S/16) tiles total
  const int bh = tile / (kS / 16);
  const int q0 = (tile % (kS / 16)) * 16;
  const __bf16* Q = Qb + (size_t)bh * kS * kDQ;
  const __bf16* K = Kb + (size_t)bh * kS * kDQ;
  const __bf16* V = Vt + (size_t)bh * kDV * kS;

  // hoist Q fragments (A layout): 6 x v16bf
  Frag16 qf[kDQC];
  const int mrow = q0 + lo;
#pragma unroll
  for (int c = 0; c < kDQC; ++c) {
    qf[c].h[0] = *(const v8bf*)(Q + (size_t)mrow * kDQ + c * 32 + 8 * hi);
    qf[c].h[1] = *(const v8bf*)(Q + (size_t)mrow * kDQ + c * 32 + 16 + 8 * hi);
  }

  float mst[8], lst[8];
  v8f o[8];
#pragma unroll
  for (int r = 0; r < 8; ++r) { mst[r] = -1e30f; lst[r] = 0.f; }
#pragma unroll
  for (int nt = 0; nt < 8; ++nt) o[nt] = v8f_zero();

  __bf16* P = &Pl[wave][0];
  const int nk = (q0 + 47) / 32;               // ceil((q0+16)/32) causal k tiles

  for (int kt = 0; kt < nk; ++kt) {
    const int k0 = kt * 32;
    if (kt + 1 < nk) {                         // hint next K/V tiles toward cache
      __builtin_prefetch(K + (size_t)(k0 + 32 + lane) * kDQ, 0, 1);
      __builtin_prefetch(V + (size_t)(lane * 4) * kS + k0 + 32, 0, 1);
    }
    v8f sc[2] = { v8f_zero(), v8f_zero() };
    // S = Q (16x192) . K^T : B-frag rows come straight from row-major K
#pragma unroll
    for (int nt = 0; nt < 2; ++nt) {
      const int krow = k0 + nt * 16 + lo;
      Frag16 kf[kDQC];
#pragma unroll
      for (int c = 0; c < kDQC; ++c) {
        kf[c].h[0] = *(const v8bf*)(K + (size_t)krow * kDQ + c * 32 + 8 * hi);
        kf[c].h[1] = *(const v8bf*)(K + (size_t)krow * kDQ + c * 32 + 16 + 8 * hi);
      }
#pragma unroll
      for (int c = 0; c < kDQC; ++c)
        sc[nt] = __builtin_amdgcn_wmma_f32_16x16x32_bf16(
            false, qf[c].v, false, kf[c].v, (short)0, sc[nt], false, false);
    }
    // online softmax over the 32 keys (rows live across 16 lanes per half)
#pragma unroll
    for (int r = 0; r < 8; ++r) {
      const int m = q0 + r + 8 * hi;
      float s0 = (k0 + lo      <= m) ? sc[0][r] : -1e30f;
      float s1 = (k0 + 16 + lo <= m) ? sc[1][r] : -1e30f;
      float mx = fmaxf(s0, s1);
#pragma unroll
      for (int off = 1; off < 16; off <<= 1) mx = fmaxf(mx, __shfl_xor(mx, off, 32));
      const float mn   = fmaxf(mst[r], mx);
      const float corr = __expf(mst[r] - mn);
      const float e0 = __expf(s0 - mn);
      const float e1 = __expf(s1 - mn);
      float rs = e0 + e1;
#pragma unroll
      for (int off = 1; off < 16; off <<= 1) rs += __shfl_xor(rs, off, 32);
      lst[r] = lst[r] * corr + rs;
      mst[r] = mn;
#pragma unroll
      for (int nt = 0; nt < 8; ++nt) o[nt][r] *= corr;
      P[(r + 8 * hi) * GLD + lo]      = (__bf16)e0;   // C-layout -> LDS [m][key]
      P[(r + 8 * hi) * GLD + 16 + lo] = (__bf16)e1;
    }
    // reload P in A-fragment layout (same-wave DS ops stay in order)
    Frag16 pf;
    pf.h[0] = *(const v8bf*)&P[lo * GLD + 8 * hi];
    pf.h[1] = *(const v8bf*)&P[lo * GLD + 16 + 8 * hi];
    // O += P (16x32) . V (32x128); V pre-transposed so B-frags are contiguous
#pragma unroll
    for (int g = 0; g < 2; ++g) {
      Frag16 vf[4];
#pragma unroll
      for (int j = 0; j < 4; ++j) {
        const int dv = (g * 4 + j) * 16 + lo;
        vf[j].h[0] = *(const v8bf*)(V + (size_t)dv * kS + k0 + 8 * hi);
        vf[j].h[1] = *(const v8bf*)(V + (size_t)dv * kS + k0 + 16 + 8 * hi);
      }
#pragma unroll
      for (int j = 0; j < 4; ++j)
        o[g * 4 + j] = __builtin_amdgcn_wmma_f32_16x16x32_bf16(
            false, pf.v, false, vf[j].v, (short)0, o[g * 4 + j], false, false);
    }
  }

  const int b = bh / kH, h = bh % kH;
#pragma unroll
  for (int nt = 0; nt < 8; ++nt) {
    const int dv = nt * 16 + lo;
#pragma unroll
    for (int r = 0; r < 8; ++r) {
      const int s = q0 + r + 8 * hi;
      Ao[((size_t)(b * kS + s)) * (kH * kDV) + h * kDV + dv] = o[nt][r] / lst[r];
    }
  }
}

// ---------------------------------------------------------------------------
// Host-side orchestration.
// ---------------------------------------------------------------------------
extern "C" void kernel_launch(void* const* d_in, const int* in_sizes, int n_in,
                              void* d_out, int out_size, void* d_ws, size_t ws_size,
                              hipStream_t stream) {
  (void)in_sizes; (void)n_in; (void)out_size; (void)ws_size;
  const float* hidden = (const float*)d_in[0];
  const float* cosb   = (const float*)d_in[1];
  const float* sinb   = (const float*)d_in[2];
  const float* Wqa    = (const float*)d_in[3];
  const float* qa_ln  = (const float*)d_in[4];
  const float* Wqb    = (const float*)d_in[5];
  const float* Wkva   = (const float*)d_in[6];
  const float* kva_ln = (const float*)d_in[7];
  const float* Wkvb   = (const float*)d_in[8];
  const float* Wo     = (const float*)d_in[9];
  float* out = (float*)d_out;

  char* ws = (char*)d_ws;
  size_t off = 0;
  auto alloc = [&](size_t bytes) -> void* {
    void* p = ws + off;
    off += (bytes + 255) & ~(size_t)255;
    return p;
  };
  const int NKV = kKVL + kDR;                                   // 576
  float*  qa   = (float*) alloc((size_t)kM * kQL * 4);          // hidden @ Wqa
  float*  qan  = (float*) alloc((size_t)kM * kQL * 4);          // rmsnorm(qa)
  float*  qpr  = (float*) alloc((size_t)kM * kH * kDQ * 4);     // qan @ Wqb
  float*  ckv  = (float*) alloc((size_t)kM * NKV * 4);          // hidden @ Wkva
  float*  ckvn = (float*) alloc((size_t)kM * kKVL * 4);         // rmsnorm(k_lat)
  float*  kvpr = (float*) alloc((size_t)kM * kH * (kDN + kDV) * 4); // ckvn @ Wkvb
  __bf16* Qb   = (__bf16*)alloc((size_t)kM * kH * kDQ * 2);     // [b,h,s,192]
  __bf16* Kb   = (__bf16*)alloc((size_t)kM * kH * kDQ * 2);     // [b,h,s,192]
  __bf16* Vt   = (__bf16*)alloc((size_t)kM * kH * kDV * 2);     // [b,h,dv,s]
  float*  attn = (float*) alloc((size_t)kM * kH * kDV * 4);     // [b,s,h*dv]

  const dim3 blk(256);
  gemm_f32_wmma_bf16<<<dim3(kQL / 128, kM / 128), blk, 0, stream>>>(
      hidden, Wqa, qa, kQL, kD);
  rmsnorm_rows<<<kM, 256, 0, stream>>>(qa, qa_ln, qan, kQL, kQL, kQL);
  gemm_f32_wmma_bf16<<<dim3((kH * kDQ) / 128, kM / 128), blk, 0, stream>>>(
      qan, Wqb, qpr, kH * kDQ, kQL);
  gemm_f32_wmma_bf16<<<dim3((NKV + 127) / 128, kM / 128), blk, 0, stream>>>(
      hidden, Wkva, ckv, NKV, kD);
  rmsnorm_rows<<<kM, 256, 0, stream>>>(ckv, kva_ln, ckvn, kKVL, NKV, kKVL);
  gemm_f32_wmma_bf16<<<dim3((kH * (kDN + kDV)) / 128, kM / 128), blk, 0, stream>>>(
      ckvn, Wkvb, kvpr, kH * (kDN + kDV), kKVL);
  prep_qkv<<<kM, 256, 0, stream>>>(qpr, kvpr, ckv, cosb, sinb, Qb, Kb, Vt);
  mla_flash<<<(kB * kH * (kS / 16)) / 8, 256, 0, stream>>>(Qb, Kb, Vt, attn);
  gemm_f32_wmma_bf16<<<dim3(kD / 128, kM / 128), blk, 0, stream>>>(
      attn, Wo, out, kM == 0 ? 0 : kD, kH * kDV);
}
